// RPN_16286515987055
// MI455X (gfx1250) — matrix operand order repaired and checked by reading, not compiled, not run on hardware
//
#include <hip/hip_runtime.h>

#define HH 64
#define WW 64
#define CIN 1024
#define CMID 512
#define NPIX 4096
#define NANCH 12288
#define NWORDS 384            // 12288 / 32
#define NCHUNK 288            // K = 9216, stepped by 32
#define WELEMS (NCHUNK*512*32) // reordered weight halves = 4,718,592

typedef __attribute__((ext_vector_type(16))) __bf16        v16bf;
typedef __attribute__((ext_vector_type(8)))  float         v8f;
typedef __attribute__((ext_vector_type(4)))  unsigned int  u32x4;
typedef __attribute__((ext_vector_type(8)))  unsigned int  u32x8;

union FragBF { v16bf v; u32x4 q[2]; };

__device__ __forceinline__ unsigned short f2bf_rne(float f) {
    unsigned int u = __float_as_uint(f);
    unsigned int r = u + 0x7FFFu + ((u >> 16) & 1u);
    return (unsigned short)(r >> 16);
}
__device__ __forceinline__ float bf2f(unsigned short h) {
    return __uint_as_float(((unsigned int)h) << 16);
}

// ---------------------------------------------------------------------------
// Kernel 0a: split fp32 input into bf16 hi/lo pair (NHWC layout preserved).
// ---------------------------------------------------------------------------
__global__ __launch_bounds__(256) void split_input(
    const float* __restrict__ in, unsigned short* __restrict__ hi,
    unsigned short* __restrict__ lo, int n)
{
    const int i = blockIdx.x * 256 + threadIdx.x;
    if (i < n) {
        const float v = in[i];
        const unsigned short h = f2bf_rne(v);
        hi[i] = h;
        lo[i] = f2bf_rne(v - bf2f(h));
    }
}

// ---------------------------------------------------------------------------
// Kernel 0b: split weights into bf16 hi/lo AND reorder into tile-major layout
// [chunk(288)][n(512)][k(32)], so each block's B tile is a contiguous 4KB blob
// fetchable by one TDM descriptor.
// ---------------------------------------------------------------------------
__global__ __launch_bounds__(256) void split_weights(
    const float* __restrict__ w, unsigned short* __restrict__ hi,
    unsigned short* __restrict__ lo)
{
    const int o = blockIdx.x * 256 + threadIdx.x;     // output half index
    if (o < WELEMS) {
        const int kk    = o & 31;
        const int n     = (o >> 5) & 511;
        const int chunk = o >> 14;
        const int tap   = chunk >> 5;
        const int c     = ((chunk & 31) << 5) + kk;
        const float v = w[((tap << 10) + c) * CMID + n];
        const unsigned short h = f2bf_rne(v);
        hi[o] = h;
        lo[o] = f2bf_rne(v - bf2f(h));
    }
}

// ---------------------------------------------------------------------------
// Kernel 1: 3x3 conv (1024 -> 512) + bias + ReLU as implicit GEMM with
// split-bf16 WMMA (Ahi*Bhi + Ahi*Blo + Alo*Bhi accumulated in f32).
// Block tile: 64 pixels (one image row) x 64 out-channels; 8 waves (2M x 4N),
// each wave owns a 32x16 output tile = two 16x16 f32 accumulators.
// B tiles arrive via the Tensor Data Mover (one tensor_load_to_lds per
// hi/lo buffer, issued by wave 0 / wave 1); A tiles via b128 copy w/ clamped
// addresses + vector select for the SAME zero-padding.
// ---------------------------------------------------------------------------
__global__ __launch_bounds__(256) void conv3x3_wmma(
    const unsigned short* __restrict__ in_hi, const unsigned short* __restrict__ in_lo,
    const unsigned short* __restrict__ w_hi,  const unsigned short* __restrict__ w_lo,
    const float* __restrict__ bias, float* __restrict__ xout)
{
    __shared__ __align__(16) unsigned short a_hi[64 * 32];
    __shared__ __align__(16) unsigned short a_lo[64 * 32];
    __shared__ __align__(16) unsigned short b_hi[64 * 32]; // [n][k]
    __shared__ __align__(16) unsigned short b_lo[64 * 32];

    const int tid    = threadIdx.x;
    const int lane   = tid & 31;
    const int waveId = tid >> 5;
    const int waveM  = waveId >> 2;       // 0..1 -> 32 rows each
    const int waveN  = waveId & 3;        // 0..3 -> 16 cols each
    const int py     = blockIdx.x;        // image row (64 blocks in M)
    const int nBase  = blockIdx.y * 64;   // out-channel base (8 blocks in N)

    v8f c0 = {}; v8f c1 = {};

    // A staging task: 64 pixels x 4 k-groups of 8 halves = 256 tasks
    const int pA  = tid >> 2;             // pixel 0..63 (= px)
    const int kgA = (tid & 3) * 8;        // k-group base

    // wave-uniform id for TDM issue (TDM ignores EXEC -> need scalar branch)
    const int wids = __builtin_amdgcn_readfirstlane(waveId);

    // TDM descriptor group 1 (constant): data_size=4B, 1-D tile of 1024 dwords,
    // tensor_dim0 = whole weight array (0x240000 dwords), tensor_dim1 = 1.
    u32x8 g1;
    g1[0] = (2u << 16);               // workgroup_mask=0, data_size=2 (4B)
    g1[1] = 0;                        // abar=0, tensor_dim0[15:0]=0
    g1[2] = (1u << 16) | 0x24u;       // tensor_dim1[15:0]=1 | tensor_dim0[31:16]
    g1[3] = (1024u << 16);            // tile_dim0=1024 dwords, tensor_dim1 hi=0
    g1[4] = 0;                        // tile_dim1=0 (1-D), tile_dim2=0
    g1[5] = 0x240000u;                // tensor_dim0_stride lo
    g1[6] = 0;                        // stride hi / dim1_stride lo
    g1[7] = 0;

    const unsigned ldsBhi = (unsigned)(size_t)(void*)b_hi;
    const unsigned ldsBlo = (unsigned)(size_t)(void*)b_lo;
    const unsigned long long wTileOff = (unsigned long long)nBase * 64ull; // bytes
    const unsigned long long gBaseHi = (unsigned long long)(size_t)w_hi + wTileOff;
    const unsigned long long gBaseLo = (unsigned long long)(size_t)w_lo + wTileOff;

    const int m16 = lane & 15, hib = lane >> 4;
    const int aOff0 = (waveM * 32 +      m16) * 32 + hib * 8;
    const int aOff1 = (waveM * 32 + 16 + m16) * 32 + hib * 8;
    const int bOff  = (waveN * 16 +      m16) * 32 + hib * 8;
    const u32x4 qzero = {0u, 0u, 0u, 0u};

    for (int chunk = 0; chunk < NCHUNK; ++chunk) {
        // ---- B tiles via Tensor Data Mover (waves 0 and 1) ----
        if (wids < 2) {
            const unsigned long long ga =
                (wids ? gBaseLo : gBaseHi) + (unsigned long long)chunk * 32768ull;
            u32x4 g0;
            g0[0] = 1u;                                   // count=1 (valid)
            g0[1] = wids ? ldsBlo : ldsBhi;               // lds_addr
            g0[2] = (unsigned)ga;                         // global_addr lo
            g0[3] = (unsigned)(ga >> 32) | (2u << 30);    // addr hi | type=2
            asm volatile("tensor_load_to_lds %0, %1" :: "s"(g0), "s"(g1) : "memory");
        }

        // ---- A tile: 64 px x 32 ch, zero-padded SAME, branch-free ----
        const int tap = chunk >> 5;            // 0..8
        const int c0k = (chunk & 31) << 5;     // channel base within tap
        const int ty  = (tap * 11) >> 5;       // tap / 3 for 0..8
        const int tx  = tap - ty * 3;
        const int iy  = py + ty - 1;
        const int ix  = pA + tx - 1;
        const bool ok = (iy >= 0) && (iy < HH) && (ix >= 0) && (ix < WW);
        const int iyc = min(max(iy, 0), HH - 1);
        const int ixc = min(max(ix, 0), WW - 1);
        const size_t off = (size_t)(iyc * WW + ixc) * CIN + c0k + kgA;
        u32x4 qh = *(const u32x4*)(in_hi + off);
        u32x4 ql = *(const u32x4*)(in_lo + off);
        qh = ok ? qh : qzero;
        ql = ok ? ql : qzero;
        *(u32x4*)(a_hi + pA * 32 + kgA) = qh;
        *(u32x4*)(a_lo + pA * 32 + kgA) = ql;

        __builtin_amdgcn_s_wait_tensorcnt(0);
        __syncthreads();

        // ---- fragments + 6 WMMAs (split-bf16: hh + hl + lh) ----
        FragBF Ah0, Al0, Ah1, Al1, Bh, Bl;
        Ah0.q[0] = *(const u32x4*)(a_hi + aOff0);
        Ah0.q[1] = *(const u32x4*)(a_hi + aOff0 + 16);
        Al0.q[0] = *(const u32x4*)(a_lo + aOff0);
        Al0.q[1] = *(const u32x4*)(a_lo + aOff0 + 16);
        Ah1.q[0] = *(const u32x4*)(a_hi + aOff1);
        Ah1.q[1] = *(const u32x4*)(a_hi + aOff1 + 16);
        Al1.q[0] = *(const u32x4*)(a_lo + aOff1);
        Al1.q[1] = *(const u32x4*)(a_lo + aOff1 + 16);
        Bh.q[0]  = *(const u32x4*)(b_hi + bOff);
        Bh.q[1]  = *(const u32x4*)(b_hi + bOff + 16);
        Bl.q[0]  = *(const u32x4*)(b_lo + bOff);
        Bl.q[1]  = *(const u32x4*)(b_lo + bOff + 16);

        c0 = __builtin_amdgcn_wmma_f32_16x16x32_bf16(false, Ah0.v, false, Bh.v, (short)0, c0, false, false);
        c0 = __builtin_amdgcn_wmma_f32_16x16x32_bf16(false, Ah0.v, false, Bl.v, (short)0, c0, false, false);
        c0 = __builtin_amdgcn_wmma_f32_16x16x32_bf16(false, Al0.v, false, Bh.v, (short)0, c0, false, false);
        c1 = __builtin_amdgcn_wmma_f32_16x16x32_bf16(false, Ah1.v, false, Bh.v, (short)0, c1, false, false);
        c1 = __builtin_amdgcn_wmma_f32_16x16x32_bf16(false, Ah1.v, false, Bl.v, (short)0, c1, false, false);
        c1 = __builtin_amdgcn_wmma_f32_16x16x32_bf16(false, Al1.v, false, Bh.v, (short)0, c1, false, false);
        __syncthreads();
    }

    // C layout: lane = N (dup in hi half), VGPR r -> M = r + 8*(lane>>4)
    const int nOut = nBase + waveN * 16 + m16;
    const float bv = bias[nOut];
    #pragma unroll
    for (int r = 0; r < 8; ++r) {
        const int px0 = waveM * 32 +      r + 8 * hib;
        const int px1 = waveM * 32 + 16 + r + 8 * hib;
        xout[(py * WW + px0) * CMID + nOut] = fmaxf(c0[r] + bv, 0.0f);
        xout[(py * WW + px1) * CMID + nOut] = fmaxf(c1[r] + bv, 0.0f);
    }
}

// ---------------------------------------------------------------------------
// Kernel 2: 1x1 cls/reg heads + softmax(2) + anchor decode.
// One wave per pixel; lanes 0..5 = cls channels, 6..17 = reg channels.
// ---------------------------------------------------------------------------
__global__ __launch_bounds__(256) void head_kernel(
    const float* __restrict__ x,
    const float* __restrict__ cls_w, const float* __restrict__ cls_b,
    const float* __restrict__ reg_w, const float* __restrict__ reg_b,
    float* __restrict__ boxes, int* __restrict__ keep0)
{
    __shared__ float dots[8][18];
    const int tid  = threadIdx.x;
    const int lane = tid & 31;
    const int wv   = tid >> 5;
    const int m    = blockIdx.x * 8 + wv;

    const float* xr = x + m * CMID;
    const float* wp; int stride; float bias;
    if (lane < 6)       { wp = cls_w + lane;       stride = 6;  bias = cls_b[lane]; }
    else if (lane < 18) { wp = reg_w + (lane - 6); stride = 12; bias = reg_b[lane - 6]; }
    else                { wp = cls_w;              stride = 6;  bias = 0.0f; }

    float acc = 0.0f;
    #pragma unroll 4
    for (int c = 0; c < CMID; ++c) acc = fmaf(xr[c], wp[c * stride], acc);

    if (lane < 18) dots[wv][lane] = acc + bias;
    __syncthreads();

    if (lane < 3) {
        const int a = lane;
        const float wsA[3] = {90.50966799187808f, 128.0f, 181.01933598375618f};
        const float hsA[3] = {181.01933598375618f, 128.0f, 90.50966799187808f};
        const int py = m >> 6, px = m & 63;
        const float cx = (px + 0.5f) * 16.0f;
        const float cy = (py + 0.5f) * 16.0f;
        const float l0 = dots[wv][2 * a], l1 = dots[wv][2 * a + 1];
        const float score = 1.0f / (1.0f + __expf(l0 - l1)); // softmax[:,1]
        const float d0 = dots[wv][6 + 4 * a + 0];
        const float d1 = dots[wv][6 + 4 * a + 1];
        const float d2 = dots[wv][6 + 4 * a + 2];
        const float d3 = dots[wv][6 + 4 * a + 3];
        const float aw = wsA[a], ah = hsA[a];
        const float xc = cx + d0 * aw;
        const float yc = cy + d1 * ah;
        const float nw = aw * __expf(d2);
        const float nh = ah * __expf(d3);
        const int bi = (m * 3 + a) * 4;
        boxes[bi + 0] = xc - 0.5f * nw;
        boxes[bi + 1] = yc - 0.5f * nh;
        boxes[bi + 2] = xc + 0.5f * nw;
        boxes[bi + 3] = yc + 0.5f * nh;
        keep0[m * 3 + a] = (score > 0.15f) ? 1 : 0;
    }
}

// ---------------------------------------------------------------------------
// Kernel 3a: parallel suppression bit-matrix  sup[i][word] (j>i, IoU>0.7).
// ---------------------------------------------------------------------------
__global__ __launch_bounds__(256) void supmat_kernel(
    const float* __restrict__ boxes, unsigned int* __restrict__ sup)
{
    const int idx = blockIdx.x * 256 + threadIdx.x;
    if (idx >= NANCH * NWORDS) return;
    const int i = idx / NWORDS;
    const int t = idx % NWORDS;
    unsigned int word = 0;
    const int jb = t * 32;
    if (jb + 31 > i) {
        const float x1 = boxes[i * 4 + 0], y1 = boxes[i * 4 + 1];
        const float x2 = boxes[i * 4 + 2], y2 = boxes[i * 4 + 3];
        const float ai = (x2 - x1) * (y2 - y1);
        #pragma unroll 4
        for (int jj = 0; jj < 32; ++jj) {
            const int j = jb + jj;
            if (j > i) {
                const float bx1 = boxes[j * 4 + 0], by1 = boxes[j * 4 + 1];
                const float bx2 = boxes[j * 4 + 2], by2 = boxes[j * 4 + 3];
                const float xx1 = fmaxf(x1, bx1), yy1 = fmaxf(y1, by1);
                const float xx2 = fminf(x2, bx2), yy2 = fminf(y2, by2);
                const float inter = fmaxf(xx2 - xx1, 0.0f) * fmaxf(yy2 - yy1, 0.0f);
                const float aj = (bx2 - bx1) * (by2 - by1);
                const float iou = inter / (ai + aj - inter + 1e-9f);
                if (iou > 0.7f) word |= (1u << jj);
            }
        }
    }
    sup[idx] = word;
}

// ---------------------------------------------------------------------------
// Kernel 3b: sequential greedy resolve (one block, keep mask in LDS),
// 32 boxes per step, one barrier per step; each thread replays the in-block
// keep-word evolution locally. Tail: out = boxes * keep.
// ---------------------------------------------------------------------------
__global__ __launch_bounds__(384) void nms_kernel(
    const unsigned int* __restrict__ sup, const int* __restrict__ keep0,
    const float* __restrict__ boxes, float* __restrict__ out)
{
    __shared__ unsigned int keepw[NWORDS];
    const int t = threadIdx.x;

    unsigned int w0 = 0;
    #pragma unroll 4
    for (int jj = 0; jj < 32; ++jj)
        if (keep0[t * 32 + jj]) w0 |= (1u << jj);
    keepw[t] = w0;
    __syncthreads();

    for (int b = 0; b < NWORDS; ++b) {
        unsigned int wb = keepw[b];            // uniform across block
        if (wb) {
            unsigned int wt = keepw[t];
            const unsigned int* rowbase = sup + (size_t)(b * 32) * NWORDS;
            for (int ii = 0; ii < 32; ++ii) {
                if ((wb >> ii) & 1u) {
                    const unsigned int* row = rowbase + (size_t)ii * NWORDS;
                    wb &= ~row[b];             // broadcast load
                    wt &= ~row[t];             // coalesced load
                }
            }
            keepw[t] = wt;
        }
        __syncthreads();
    }

    const unsigned int wt = keepw[t];
    #pragma unroll 4
    for (int jj = 0; jj < 32; ++jj) {
        const int j = t * 32 + jj;
        const float k = ((wt >> jj) & 1u) ? 1.0f : 0.0f;
        out[j * 4 + 0] = boxes[j * 4 + 0] * k;
        out[j * 4 + 1] = boxes[j * 4 + 1] * k;
        out[j * 4 + 2] = boxes[j * 4 + 2] * k;
        out[j * 4 + 3] = boxes[j * 4 + 3] * k;
    }
}

// ---------------------------------------------------------------------------
extern "C" void kernel_launch(void* const* d_in, const int* in_sizes, int n_in,
                              void* d_out, int out_size, void* d_ws, size_t ws_size,
                              hipStream_t stream)
{
    const float* in     = (const float*)d_in[0];
    const float* conv_w = (const float*)d_in[1];
    const float* conv_b = (const float*)d_in[2];
    const float* cls_w  = (const float*)d_in[3];
    const float* cls_b  = (const float*)d_in[4];
    const float* reg_w  = (const float*)d_in[5];
    const float* reg_b  = (const float*)d_in[6];
    float* out = (float*)d_out;

    // workspace layout (~44.3 MB):
    float* x      = (float*)d_ws;                     //  4096*512 f32   (8.39 MB)
    float* boxes  = x + NPIX * CMID;                  // 12288*4  f32
    int*   keep0  = (int*)(boxes + NANCH * 4);        // 12288    i32
    unsigned short* in_hi = (unsigned short*)(keep0 + NANCH);   // bf16 region...
    unsigned short* in_lo = in_hi + NPIX * CIN;
    unsigned short* w_hi  = in_lo + NPIX * CIN;
    unsigned short* w_lo  = w_hi + WELEMS;
    // sup matrix reuses the bf16 region (dead after conv3x3_wmma)
    unsigned int* sup = (unsigned int*)in_hi;         // 12288*384 u32  (18.9 MB)

    const int nIn = NPIX * CIN;
    split_input  <<<dim3((nIn + 255) / 256), 256, 0, stream>>>(in, in_hi, in_lo, nIn);
    split_weights<<<dim3((WELEMS + 255) / 256), 256, 0, stream>>>(conv_w, w_hi, w_lo);
    conv3x3_wmma <<<dim3(64, 8), 256, 0, stream>>>(in_hi, in_lo, w_hi, w_lo, conv_b, x);
    head_kernel  <<<dim3(512), 256, 0, stream>>>(x, cls_w, cls_b, reg_w, reg_b, boxes, keep0);
    supmat_kernel<<<dim3((NANCH * NWORDS + 255) / 256), 256, 0, stream>>>(boxes, sup);
    nms_kernel   <<<dim3(1), 384, 0, stream>>>(sup, keep0, boxes, out);
}
